// QNet_51754355917331
// MI455X (gfx1250) — matrix operand order repaired and checked by reading, not compiled, not run on hardware
//
#include <hip/hip_runtime.h>

// Problem constants (from reference)
#define B_TOT 8192
#define A_AG  32
#define NOBS  128
#define H1    128
#define HX    64
#define NACT  16
#define BM    128      // batch rows per workgroup
#define LDK   136      // padded LDS stride for K=128 rows (f16 elems)
#define LDK64 72       // padded LDS stride for K=64 rows

typedef __attribute__((ext_vector_type(16))) _Float16 v16h;
typedef __attribute__((ext_vector_type(8)))  float    v8f;

// ---- WMMA fragment loaders (wave32, 16x16x32 f16) -------------------------
// A operand: S row-major [M][ld]; rows row..row+15, cols kbase..kbase+31.
// Layout used here: lane = m + 16*half ; per-lane K = half*16 + e  (contiguous)
__device__ __forceinline__ v16h frag_a_rowmajor(const _Float16* S, int ld,
                                                int row, int kbase) {
  int lane = threadIdx.x & 31;
  int m = lane & 15, half = lane >> 4;
  const _Float16* p = S + (row + m) * ld + kbase + half * 16;
  v16h f;
#pragma unroll
  for (int e = 0; e < 16; ++e) f[e] = p[e];
  return f;
}

// B operand: S stored COLUMN-major [N][ld] (i.e. S[n*ld + k]); block
// k = kbase..kbase+31, n = ncol..ncol+15.
// Layout used here: lane = n + 16*half ; per-lane K = half*16 + e  (contiguous 32B)
__device__ __forceinline__ v16h frag_b_colmajor(const _Float16* S, int ld,
                                                int kbase, int ncol) {
  int lane = threadIdx.x & 31;
  int n = lane & 15, half = lane >> 4;
  const _Float16* p = S + (ncol + n) * ld + kbase + half * 16;
  v16h f;
#pragma unroll
  for (int e = 0; e < 16; ++e) f[e] = p[e];
  return f;
}

__device__ __forceinline__ v8f wmma_f16(v16h a, v16h b, v8f c) {
  return __builtin_amdgcn_wmma_f32_16x16x32_f16(false, a, false, b,
                                                (short)0, c, false, false);
}

__device__ __forceinline__ v8f zero8() {
  v8f z = {0.f, 0.f, 0.f, 0.f, 0.f, 0.f, 0.f, 0.f};
  return z;
}

// ---- Kernel 1: per-agent encoder  h = relu(relu(obs@W1+b1)@W2+b2) ---------
__global__ __launch_bounds__(256) void qnet_encode(
    const float* __restrict__ obs, const float* __restrict__ W1,
    const float* __restrict__ b1, const float* __restrict__ W2,
    const float* __restrict__ b2, _Float16* __restrict__ h_out) {
  __shared__ _Float16 sO[BM * LDK];    // obs tile, row-major [m][k]
  __shared__ _Float16 sW1[H1 * LDK];   // W1, col-major [n][k]
  __shared__ _Float16 sX[BM * LDK];    // x tile, row-major [m][k]
  __shared__ _Float16 sW2[HX * LDK];   // W2, col-major [n][k]

  const int tid = threadIdx.x;
  const int a   = blockIdx.y;
  const int b0  = blockIdx.x * BM;

  // Stage obs tile (fp32 -> f16)
  for (int idx = tid; idx < BM * NOBS; idx += 256) {
    int m = idx >> 7, k = idx & 127;
    sO[m * LDK + k] =
        (_Float16)obs[((size_t)(b0 + m) * A_AG + a) * NOBS + k];
  }
  // Stage W1[a] transposed: global [k][n] -> LDS [n][k]
  const float* W1a = W1 + (size_t)a * NOBS * H1;
  for (int idx = tid; idx < NOBS * H1; idx += 256) {
    int k = idx >> 7, n = idx & 127;
    sW1[n * LDK + k] = (_Float16)W1a[idx];
  }
  // Stage W2[a] transposed: global [k][n(64)] -> LDS [n][k]
  const float* W2a = W2 + (size_t)a * H1 * HX;
  for (int idx = tid; idx < H1 * HX; idx += 256) {
    int k = idx >> 6, n = idx & 63;
    sW2[n * LDK + k] = (_Float16)W2a[idx];
  }
  __syncthreads();

  const int wave = tid >> 5, lane = tid & 31;
  const int lm = lane & 15, lh = lane >> 4;
  const int m0 = wave * 16;  // each wave owns a 16-row strip

  // GEMM1: x[128x128] = obs @ W1
  v8f acc[8];
#pragma unroll
  for (int nt = 0; nt < 8; ++nt) acc[nt] = zero8();
#pragma unroll
  for (int kk = 0; kk < 4; ++kk) {
    v16h af = frag_a_rowmajor(sO, LDK, m0, kk * 32);
#pragma unroll
    for (int nt = 0; nt < 8; ++nt) {
      v16h bfr = frag_b_colmajor(sW1, LDK, kk * 32, nt * 16);
      acc[nt] = wmma_f16(af, bfr, acc[nt]);
    }
  }
  // epilogue: +b1, relu, write f16 x tile (own rows only -> no barrier)
  const float* b1a = b1 + (size_t)a * H1;
#pragma unroll
  for (int nt = 0; nt < 8; ++nt) {
    int n = nt * 16 + lm;
    float bias = b1a[n];
#pragma unroll
    for (int r = 0; r < 8; ++r) {
      int m = m0 + r + 8 * lh;
      float v = acc[nt][r] + bias;
      sX[m * LDK + n] = (_Float16)(v > 0.f ? v : 0.f);
    }
  }

  // GEMM2: h[128x64] = x @ W2
  v8f acc2[4];
#pragma unroll
  for (int nt = 0; nt < 4; ++nt) acc2[nt] = zero8();
#pragma unroll
  for (int kk = 0; kk < 4; ++kk) {
    v16h af = frag_a_rowmajor(sX, LDK, m0, kk * 32);
#pragma unroll
    for (int nt = 0; nt < 4; ++nt) {
      v16h bfr = frag_b_colmajor(sW2, LDK, kk * 32, nt * 16);
      acc2[nt] = wmma_f16(af, bfr, acc2[nt]);
    }
  }
  const float* b2a = b2 + (size_t)a * HX;
#pragma unroll
  for (int nt = 0; nt < 4; ++nt) {
    int n = nt * 16 + lm;
    float bias = b2a[n];
#pragma unroll
    for (int r = 0; r < 8; ++r) {
      int m = m0 + r + 8 * lh;
      float v = acc2[nt][r] + bias;
      h_out[((size_t)(b0 + m) * A_AG + a) * HX + n] =
          (_Float16)(v > 0.f ? v : 0.f);
    }
  }
}

// ---- Kernel 2: comm mix + decoder  q = tanh([h|comm]@Wc+bc)@Wd + bd -------
__global__ __launch_bounds__(256) void qnet_decode(
    const _Float16* __restrict__ h, const float* __restrict__ Wc,
    const float* __restrict__ bc, const float* __restrict__ Wd,
    const float* __restrict__ bd, const int* __restrict__ comm_mask,
    float* __restrict__ qout) {
  __shared__ _Float16 sCI[BM * LDK];     // [m][0:64]=h_i, [64:128]=comm
  __shared__ _Float16 sWc[HX * LDK];     // Wc col-major [n][k=128]
  __shared__ _Float16 sH2[BM * LDK64];   // h2 tile [m][k=64]
  __shared__ _Float16 sWd[NACT * LDK64]; // Wd col-major [n=16][k=64]
  __shared__ float sWrow[A_AG];          // normalized comm weights w[i][j]

  const int tid = threadIdx.x;
  const int i   = blockIdx.y;   // receiving agent
  const int b0  = blockIdx.x * BM;

  if (tid < A_AG) {
    float cnt = 0.f;
    for (int j = 0; j < A_AG; ++j)
      cnt += (comm_mask[i * A_AG + j] != 0 && j != i) ? 1.f : 0.f;
    float mv = (comm_mask[i * A_AG + tid] != 0 && tid != i) ? 1.f : 0.f;
    sWrow[tid] = mv / fmaxf(cnt, 1.f);
  }
  // Stage Wc[i], Wd[i] transposed to col-major f16
  const float* Wci = Wc + (size_t)i * (2 * HX) * HX;
  for (int idx = tid; idx < 2 * HX * HX; idx += 256) {
    int k = idx >> 6, n = idx & 63;
    sWc[n * LDK + k] = (_Float16)Wci[idx];
  }
  const float* Wdi = Wd + (size_t)i * HX * NACT;
  for (int idx = tid; idx < HX * NACT; idx += 256) {
    int k = idx >> 4, n = idx & 15;
    sWd[n * LDK64 + k] = (_Float16)Wdi[idx];
  }
  // Own hidden state -> cols [0,64)
  for (int idx = tid; idx < BM * HX; idx += 256) {
    int m = idx >> 6, c = idx & 63;
    sCI[m * LDK + c] = h[((size_t)(b0 + m) * A_AG + i) * HX + c];
  }
  __syncthreads();  // sWrow ready

  // comm = sum_j w[i][j] * h[b][j][:]  -> cols [64,128). h is L2-resident.
  for (int idx = tid; idx < BM * HX; idx += 256) {
    int m = idx >> 6, c = idx & 63;
    const _Float16* hp = h + (size_t)(b0 + m) * A_AG * HX + c;
    float s = 0.f;
#pragma unroll 4
    for (int j = 0; j < A_AG; ++j) s += sWrow[j] * (float)hp[j * HX];
    sCI[m * LDK + 64 + c] = (_Float16)s;
  }
  __syncthreads();

  const int wave = tid >> 5, lane = tid & 31;
  const int lm = lane & 15, lh = lane >> 4;
  const int m0 = wave * 16;

  // GEMM3: h2[128x64] = tanh(comm_in @ Wc + bc)
  v8f acc[4];
#pragma unroll
  for (int nt = 0; nt < 4; ++nt) acc[nt] = zero8();
#pragma unroll
  for (int kk = 0; kk < 4; ++kk) {
    v16h af = frag_a_rowmajor(sCI, LDK, m0, kk * 32);
#pragma unroll
    for (int nt = 0; nt < 4; ++nt) {
      v16h bfr = frag_b_colmajor(sWc, LDK, kk * 32, nt * 16);
      acc[nt] = wmma_f16(af, bfr, acc[nt]);
    }
  }
  const float* bci = bc + (size_t)i * HX;
#pragma unroll
  for (int nt = 0; nt < 4; ++nt) {
    int n = nt * 16 + lm;
    float bias = bci[n];
#pragma unroll
    for (int r = 0; r < 8; ++r) {
      int m = m0 + r + 8 * lh;
      sH2[m * LDK64 + n] = (_Float16)tanhf(acc[nt][r] + bias);
    }
  }

  // GEMM4: q[128x16] = h2 @ Wd + bd  (single N-tile, K=64 -> 2 k-steps)
  v8f accq = zero8();
#pragma unroll
  for (int kk = 0; kk < 2; ++kk) {
    v16h af  = frag_a_rowmajor(sH2, LDK64, m0, kk * 32);
    v16h bfr = frag_b_colmajor(sWd, LDK64, kk * 32, 0);
    accq = wmma_f16(af, bfr, accq);
  }
  float bias = bd[(size_t)i * NACT + lm];
#pragma unroll
  for (int r = 0; r < 8; ++r) {
    int m = m0 + r + 8 * lh;
    qout[((size_t)(b0 + m) * A_AG + i) * NACT + lm] = accq[r] + bias;
  }
}

extern "C" void kernel_launch(void* const* d_in, const int* in_sizes, int n_in,
                              void* d_out, int out_size, void* d_ws,
                              size_t ws_size, hipStream_t stream) {
  (void)in_sizes; (void)n_in; (void)out_size; (void)ws_size;
  const float* obs = (const float*)d_in[0];
  const float* W1  = (const float*)d_in[1];
  const float* b1  = (const float*)d_in[2];
  const float* W2  = (const float*)d_in[3];
  const float* b2  = (const float*)d_in[4];
  // d_in[5] = Wg, d_in[6] = bg : dead in the reference (gates unused)
  const float* Wc  = (const float*)d_in[7];
  const float* bc  = (const float*)d_in[8];
  const float* Wd  = (const float*)d_in[9];
  const float* bd  = (const float*)d_in[10];
  const int* comm_mask = (const int*)d_in[11];

  _Float16* hbuf = (_Float16*)d_ws;  // B*A*HX f16 = 32 MB staging
  float* out = (float*)d_out;

  dim3 grid(B_TOT / BM, A_AG);
  qnet_encode<<<grid, 256, 0, stream>>>(obs, W1, b1, W2, b2, hbuf);
  qnet_decode<<<grid, 256, 0, stream>>>(hbuf, Wc, bc, Wd, bd, comm_mask, out);
}
